// ThreeExpertsModel_41927470743949
// MI455X (gfx1250) — matrix operand order repaired and checked by reading, not compiled, not run on hardware
//
#include <hip/hip_runtime.h>
#include <math.h>

#define B_   32
#define S_   512
#define H_   12
#define E_   64
#define D_   768
#define P_   20
#define L_   12
#define NE_  9
#define KP_  544            // P + S padded to multiple of 32 (34 key tiles of 16)
#define M_   (B_*S_)        // 16384 rows
#define FF_  3072
#define QKV_ 2304

typedef __bf16 bf16_t;
typedef bf16_t v16bf __attribute__((ext_vector_type(16)));
typedef bf16_t v8bf  __attribute__((ext_vector_type(8)));
typedef float  v8f   __attribute__((ext_vector_type(8)));

// ---------- helpers ----------
__device__ __forceinline__ unsigned short f2bf(float f) {
    unsigned u = __float_as_uint(f);
    unsigned r = u + 0x7FFFu + ((u >> 16) & 1u);   // round-to-nearest-even
    return (unsigned short)(r >> 16);
}

__device__ __forceinline__ v16bf frag2(const unsigned short* lo, const unsigned short* hi) {
    v8bf a = *reinterpret_cast<const v8bf*>(lo);
    v8bf b = *reinterpret_cast<const v8bf*>(hi);
    return __builtin_shufflevector(a, b, 0,1,2,3,4,5,6,7,8,9,10,11,12,13,14,15);
}
__device__ __forceinline__ v16bf frag16(const unsigned short* p) { return frag2(p, p + 8); }

__device__ __forceinline__ v8f wmma_bf16(v16bf a, v16bf b, v8f c) {
    // D = A(16x32) * B(32x16) + C, fp32 accumulate
    return __builtin_amdgcn_wmma_f32_16x16x32_bf16(false, a, false, b, (short)0, c, false, false);
}

// ---------- generic GEMM: C[M,N] f32 = A[M,K] bf16 * B^T (BT[N,K] bf16) ----------
// grid = (N/64, M/128), block = 256 (8 waves); wave w handles rows m0+16w..+15, 4 N-tiles
__global__ __launch_bounds__(256)
void gemm_bf16_kernel(const unsigned short* __restrict__ A,
                      const unsigned short* __restrict__ BT,
                      float* __restrict__ C, int N, int K)
{
    const int wave = threadIdx.x >> 5;
    const int lane = threadIdx.x & 31;
    const int lh = lane & 15, g = lane >> 4;
    const int m0 = blockIdx.y * 128 + wave * 16;
    const int n0 = blockIdx.x * 64;

    const unsigned short* arow = A + (size_t)(m0 + lh) * K;
    v8f acc0 = {}, acc1 = {}, acc2 = {}, acc3 = {};
    for (int kk = 0; kk < K; kk += 32) {
        // A 16x32 fragment: lane: M=lh, K = kk + g*16 .. +15 (contiguous)
        v16bf a = frag16(arow + kk + g * 16);
        // B 32x16 fragment per N-tile: lane: N = lh, K = kk + g*16 .. +15 (contiguous in BT row)
        const unsigned short* bbase = BT + (size_t)(n0 + lh) * K + kk + g * 16;
        acc0 = wmma_bf16(a, frag16(bbase), acc0);
        acc1 = wmma_bf16(a, frag16(bbase + (size_t)16 * K), acc1);
        acc2 = wmma_bf16(a, frag16(bbase + (size_t)32 * K), acc2);
        acc3 = wmma_bf16(a, frag16(bbase + (size_t)48 * K), acc3);
    }
    // C layout: VGPR r -> row m0 + r + 8*g ; col n0 + tile*16 + lh
    size_t base = (size_t)(m0 + g * 8) * N + n0 + lh;
#pragma unroll
    for (int r = 0; r < 8; ++r) {
        C[base + (size_t)r * N +  0] = acc0[r];
        C[base + (size_t)r * N + 16] = acc1[r];
        C[base + (size_t)r * N + 32] = acc2[r];
        C[base + (size_t)r * N + 48] = acc3[r];
    }
}

// ---------- fused attention: one wave per (b, h, 16-query tile) ----------
__global__ __launch_bounds__(32)
void attn_kernel(const unsigned short* __restrict__ Qm,   // [B,H,S,E] bf16
                 const unsigned short* __restrict__ Kf,   // [B,H,KP,E] bf16
                 const unsigned short* __restrict__ Vt,   // [B,H,E,KP] bf16
                 const float* __restrict__ biasb,         // [B,KP] f32
                 unsigned short* __restrict__ ctx)        // [B,S,D] bf16 (head-major cols)
{
    __shared__ float          sc[16][KP_];
    __shared__ unsigned short pb[16][KP_];
    const int b = blockIdx.z, h = blockIdx.y, s0 = blockIdx.x * 16;
    const int lane = threadIdx.x;
    const int lh = lane & 15, g = lane >> 4;
    const size_t bh = (size_t)b * H_ + h;

    const unsigned short* qrow = Qm + (bh * S_ + s0 + lh) * E_;
    v16bf a0 = frag16(qrow + g * 16);        // E 0..31
    v16bf a1 = frag16(qrow + 32 + g * 16);   // E 32..63

    const unsigned short* kbase = Kf + bh * (size_t)KP_ * E_;
    const float scale = 0.125f;   // 1/sqrt(64)
    for (int t = 0; t < KP_ / 16; ++t) {
        const unsigned short* krow = kbase + (size_t)(t * 16 + lh) * E_;
        v8f c = {};
        c = wmma_bf16(a0, frag16(krow + g * 16), c);
        c = wmma_bf16(a1, frag16(krow + 32 + g * 16), c);
#pragma unroll
        for (int r = 0; r < 8; ++r)
            sc[r + g * 8][t * 16 + lh] = c[r] * scale;
    }
    __syncthreads();

    if (lane < 16) {   // one lane per query row: exact softmax over 544 (pads get -1e30 bias)
        const float* brow = biasb + (size_t)b * KP_;
        float mx = -3.0e38f;
        for (int k = 0; k < KP_; ++k) {
            float v = sc[lane][k] + brow[k];
            sc[lane][k] = v;
            mx = fmaxf(mx, v);
        }
        float sum = 0.f;
        for (int k = 0; k < KP_; ++k) { float e = expf(sc[lane][k] - mx); sc[lane][k] = e; sum += e; }
        float inv = 1.0f / sum;
        for (int k = 0; k < KP_; ++k) pb[lane][k] = f2bf(sc[lane][k] * inv);
    }
    __syncthreads();

    v8f o0 = {}, o1 = {}, o2 = {}, o3 = {};
    const unsigned short* vbase = Vt + bh * (size_t)E_ * KP_;
    for (int c = 0; c < KP_ / 32; ++c) {
        int k0 = c * 32;
        v16bf pa = frag16(&pb[lh][k0 + g * 16]);
        const unsigned short* vr = vbase + (size_t)lh * KP_ + k0 + g * 16;
        o0 = wmma_bf16(pa, frag16(vr), o0);
        o1 = wmma_bf16(pa, frag16(vr + (size_t)16 * KP_), o1);
        o2 = wmma_bf16(pa, frag16(vr + (size_t)32 * KP_), o2);
        o3 = wmma_bf16(pa, frag16(vr + (size_t)48 * KP_), o3);
    }
    size_t obase = ((size_t)b * S_ + s0 + g * 8) * D_ + h * E_ + lh;
#pragma unroll
    for (int r = 0; r < 8; ++r) {
        ctx[obase + (size_t)r * D_ +  0] = f2bf(o0[r]);
        ctx[obase + (size_t)r * D_ + 16] = f2bf(o1[r]);
        ctx[obase + (size_t)r * D_ + 32] = f2bf(o2[r]);
        ctx[obase + (size_t)r * D_ + 48] = f2bf(o3[r]);
    }
}

// ---------- elementwise / small kernels ----------
__global__ void init_kernel(float* moe_acc) { if (threadIdx.x == 0) *moe_acc = 0.f; }

__global__ void emb_kernel(const int* __restrict__ ids, const float* __restrict__ we,
                           const float* __restrict__ pe, const float* __restrict__ te,
                           float* __restrict__ y)
{
    size_t idx = (size_t)blockIdx.x * 256 + threadIdx.x;
    if (idx >= (size_t)M_ * D_) return;
    int d = (int)(idx % D_);
    size_t row = idx / D_;
    int s = (int)(row % S_);
    int wid = ids[row];
    y[idx] = we[(size_t)wid * D_ + d] + pe[(size_t)s * D_ + d] + te[d];
}

// out = LN(xin [+ res] [+ colbias]); writes f32 + bf16 copies. One block per row, D=768.
__global__ __launch_bounds__(256)
void ln_row_kernel(const float* __restrict__ xin, const float* __restrict__ res,
                   const float* __restrict__ colbias, const float* __restrict__ gg,
                   const float* __restrict__ bb, float* __restrict__ xout,
                   unsigned short* __restrict__ xbout)
{
    __shared__ float s1[256], s2[256];
    const int row = blockIdx.x, tid = threadIdx.x;
    float v[3]; float sum = 0.f, sq = 0.f;
#pragma unroll
    for (int i = 0; i < 3; ++i) {
        int d = tid + i * 256;
        float x = xin[(size_t)row * D_ + d];
        if (res)     x += res[(size_t)row * D_ + d];
        if (colbias) x += colbias[d];
        v[i] = x; sum += x; sq += x * x;
    }
    s1[tid] = sum; s2[tid] = sq; __syncthreads();
    for (int off = 128; off > 0; off >>= 1) {
        if (tid < off) { s1[tid] += s1[tid + off]; s2[tid] += s2[tid + off]; }
        __syncthreads();
    }
    float mean = s1[0] * (1.0f / D_);
    float var  = s2[0] * (1.0f / D_) - mean * mean;
    float inv  = rsqrtf(var + 1e-12f);
#pragma unroll
    for (int i = 0; i < 3; ++i) {
        int d = tid + i * 256;
        float o = (v[i] - mean) * inv * gg[d] + bb[d];
        xout[(size_t)row * D_ + d] = o;
        xbout[(size_t)row * D_ + d] = f2bf(o);
    }
}

__global__ void bias_kernel(const float* __restrict__ amask, float* __restrict__ biasb)
{
    int idx = blockIdx.x * 256 + threadIdx.x;
    if (idx >= B_ * KP_) return;
    int k = idx % KP_, b = idx / KP_;
    float v;
    if (k < P_)            v = 0.f;
    else if (k < P_ + S_)  v = (1.0f - amask[(size_t)b * S_ + (k - P_)]) * -10000.0f;
    else                   v = -1.0e30f;   // padded keys
    biasb[idx] = v;
}

__global__ void cvtT_kernel(const float* __restrict__ W, unsigned short* __restrict__ WT,
                            int K, int N)   // W[K,N] -> WT[N,K] bf16
{
    size_t idx = (size_t)blockIdx.x * 256 + threadIdx.x;
    if (idx >= (size_t)K * N) return;
    int n = (int)(idx % N); int k = (int)(idx / N);
    WT[(size_t)n * K + k] = f2bf(W[idx]);
}

__global__ void gate_kernel(const float* __restrict__ EE, const float* __restrict__ gw,
                            const float* __restrict__ gb, int l, float* __restrict__ glog)
{
    __shared__ float red[256];
    const int n = blockIdx.x, tid = threadIdx.x;
    float acc = 0.f;
    for (int i = tid; i < P_ * D_; i += 256) {
        int p = i / D_, d = i - p * D_;
        float s = 0.f;
        for (int e = 0; e < NE_; ++e)
            s += EE[((size_t)e * P_ + p) * (L_ * 2 * D_) + (size_t)(2 * l) * D_ + d];
        acc += s * gw[((size_t)l * (P_ * D_) + i) * NE_ + n];
    }
    red[tid] = acc; __syncthreads();
    for (int off = 128; off > 0; off >>= 1) { if (tid < off) red[tid] += red[tid + off]; __syncthreads(); }
    if (tid == 0) glog[n] = red[0] + gb[l * NE_ + n];
}

__global__ void gate_softmax_kernel(const float* __restrict__ glog, float* __restrict__ w,
                                    float* __restrict__ moe_acc)
{
    if (threadIdx.x != 0) return;
    float v[NE_]; float mx = -3e38f;
    for (int i = 0; i < NE_; ++i) { v[i] = glog[i]; mx = fmaxf(mx, v[i]); }
    float s = 0.f;
    for (int i = 0; i < NE_; ++i) { v[i] = expf(v[i] - mx); s += v[i]; }
    float inv = 1.0f / s, q = 0.f;
    for (int i = 0; i < NE_; ++i) { float wi = v[i] * inv; w[i] = wi; q += wi * wi; }
    *moe_acc += q;
}

__global__ void prefix_kernel(const float* __restrict__ EE, const float* __restrict__ w,
                              int l, float* __restrict__ ck, float* __restrict__ cvt)
{
    int idx = blockIdx.x * 256 + threadIdx.x;
    if (idx >= P_ * D_) return;
    int p = idx / D_, c = idx - p * D_;
    float sk = 0.f, sv = 0.f;
    for (int e = 0; e < NE_; ++e) {
        size_t base = ((size_t)e * P_ + p) * (L_ * 2 * D_) + (size_t)(2 * l) * D_;
        float we = w[e];
        sk += we * EE[base + c];
        sv += we * EE[base + D_ + c];
    }
    ck[idx] = sk; cvt[idx] = sv;
}

__global__ void proj_kernel(const float* __restrict__ cvt, const float* __restrict__ pw,
                            const float* __restrict__ pbias, int l, float* __restrict__ cv)
{
    int idx = blockIdx.x * 256 + threadIdx.x;
    if (idx >= P_ * D_) return;
    int p = idx / D_, d = idx - p * D_;
    const float* wcol = pw + (size_t)l * D_ * D_ + d;
    const float* r = cvt + (size_t)p * D_;
    float s = pbias[(size_t)l * D_ + d];
    for (int k = 0; k < D_; ++k) s += r[k] * wcol[(size_t)k * D_];
    cv[idx] = s;
}

// write prefix rows [0,P) and zero pad rows [P+S, KP) into Kf / Vt for all (b,h)
__global__ void prefix_write_kernel(const float* __restrict__ ck, const float* __restrict__ cv,
                                    unsigned short* __restrict__ Kf, unsigned short* __restrict__ Vt)
{
    int idx = blockIdx.x * 256 + threadIdx.x;
    if (idx >= B_ * H_ * E_ * 32) return;
    int j = idx & 31; int rem = idx >> 5;
    int e = rem & (E_ - 1); rem >>= 6;
    int h = rem % H_; int b = rem / H_;
    int key = (j < P_) ? j : (S_ + j);     // j in [P,32) -> pads 532..543
    unsigned short kv = 0, vv = 0;
    if (j < P_) {
        kv = f2bf(ck[(size_t)j * D_ + h * E_ + e]);
        vv = f2bf(cv[(size_t)j * D_ + h * E_ + e]);
    }
    size_t bh = (size_t)b * H_ + h;
    Kf[(bh * KP_ + key) * E_ + e] = kv;
    Vt[(bh * (size_t)E_ + e) * KP_ + key] = vv;
}

__global__ void pack_qkv_kernel(const float* __restrict__ y, const float* __restrict__ qb, int l,
                                unsigned short* __restrict__ Qm, unsigned short* __restrict__ Kf,
                                unsigned short* __restrict__ Vt)
{
    size_t idx = (size_t)blockIdx.x * 256 + threadIdx.x;
    if (idx >= (size_t)M_ * D_) return;
    int col = (int)(idx % D_);
    size_t row = idx / D_;
    int h = col >> 6, e = col & 63;
    int b = (int)(row / S_), s = (int)(row % S_);
    const float* yr = y + row * QKV_;
    const float* qbl = qb + (size_t)l * QKV_;
    float q = yr[col]           + qbl[col];
    float k = yr[D_ + col]      + qbl[D_ + col];
    float v = yr[2 * D_ + col]  + qbl[2 * D_ + col];
    size_t bh = (size_t)b * H_ + h;
    Qm[(bh * S_ + s) * E_ + e] = f2bf(q);
    Kf[(bh * KP_ + P_ + s) * E_ + e] = f2bf(k);
    Vt[(bh * (size_t)E_ + e) * KP_ + P_ + s] = f2bf(v);
}

__global__ void gelu_kernel(const float* __restrict__ y, const float* __restrict__ fb, int l,
                            unsigned short* __restrict__ yb)
{
    size_t idx = (size_t)blockIdx.x * 256 + threadIdx.x;
    if (idx >= (size_t)M_ * FF_) return;
    int col = (int)(idx % FF_);
    float v = y[idx] + fb[(size_t)l * FF_ + col];
    float gval = 0.5f * v * (1.0f + erff(v * 0.70710678118654752f));
    yb[idx] = f2bf(gval);
}

__global__ void pool_kernel(const float* __restrict__ x, const float* __restrict__ pw,
                            const float* __restrict__ pb, float* __restrict__ pooled)
{
    int idx = blockIdx.x * 256 + threadIdx.x;
    if (idx >= B_ * D_) return;
    int d = idx % D_, b = idx / D_;
    const float* xr = x + (size_t)b * S_ * D_;   // token 0 of batch b
    float s = pb[d];
    for (int k = 0; k < D_; ++k) s += xr[k] * pw[(size_t)k * D_ + d];
    pooled[idx] = tanhf(s);
}

__global__ void logits_kernel(const float* __restrict__ pooled, const float* __restrict__ fw,
                              const float* __restrict__ fb, const float* __restrict__ moe_acc,
                              float* __restrict__ out)
{
    int t = threadIdx.x;
    if (t < 64) {
        int b = t >> 1, c = t & 1;
        const float* pr = pooled + (size_t)b * D_;
        float s = fb[c];
        for (int k = 0; k < D_; ++k) s += pr[k] * fw[k * 2 + c];
        out[t] = s;
    }
    if (t == 0) out[64] = moe_acc[0] / (float)L_;
}

// ---------- launch ----------
extern "C" void kernel_launch(void* const* d_in, const int* in_sizes, int n_in,
                              void* d_out, int out_size, void* d_ws, size_t ws_size,
                              hipStream_t stream)
{
    (void)in_sizes; (void)n_in; (void)out_size; (void)ws_size;
    const int*   ids   = (const int*)  d_in[0];
    const float* amask = (const float*)d_in[1];
    const float* EE    = (const float*)d_in[2];
    const float* gw    = (const float*)d_in[3];
    const float* gb    = (const float*)d_in[4];
    const float* pw    = (const float*)d_in[5];
    const float* pbias = (const float*)d_in[6];
    const float* wemb  = (const float*)d_in[7];
    const float* pemb  = (const float*)d_in[8];
    const float* temb  = (const float*)d_in[9];
    const float* elng  = (const float*)d_in[10];
    const float* elnb  = (const float*)d_in[11];
    const float* qkvw  = (const float*)d_in[12];
    const float* qkvb  = (const float*)d_in[13];
    const float* aow   = (const float*)d_in[14];
    const float* aob   = (const float*)d_in[15];
    const float* ln1g  = (const float*)d_in[16];
    const float* ln1b  = (const float*)d_in[17];
    const float* f1w   = (const float*)d_in[18];
    const float* f1b   = (const float*)d_in[19];
    const float* f2w   = (const float*)d_in[20];
    const float* f2b   = (const float*)d_in[21];
    const float* ln2g  = (const float*)d_in[22];
    const float* ln2b  = (const float*)d_in[23];
    const float* poolw = (const float*)d_in[24];
    const float* poolb = (const float*)d_in[25];
    const float* fcw   = (const float*)d_in[26];
    const float* fcb   = (const float*)d_in[27];
    float* out = (float*)d_out;

    char* cur = (char*)d_ws;
    auto alloc = [&](size_t bytes) -> char* {
        char* p = cur; cur += (bytes + 255) & ~(size_t)255; return p;
    };
    float*          x    = (float*)         alloc((size_t)M_ * D_ * 4);
    unsigned short* xb   = (unsigned short*)alloc((size_t)M_ * D_ * 2);
    float*          y    = (float*)         alloc((size_t)M_ * FF_ * 4);
    unsigned short* yb   = (unsigned short*)alloc((size_t)M_ * FF_ * 2);
    unsigned short* Qm   = (unsigned short*)alloc((size_t)B_ * H_ * S_ * E_ * 2);
    unsigned short* Kf   = (unsigned short*)alloc((size_t)B_ * H_ * KP_ * E_ * 2);
    unsigned short* Vt   = (unsigned short*)alloc((size_t)B_ * H_ * E_ * KP_ * 2);
    unsigned short* ctx  = (unsigned short*)alloc((size_t)M_ * D_ * 2);
    unsigned short* wqT  = (unsigned short*)alloc((size_t)D_ * QKV_ * 2);
    unsigned short* waoT = (unsigned short*)alloc((size_t)D_ * D_ * 2);
    unsigned short* wf1T = (unsigned short*)alloc((size_t)D_ * FF_ * 2);
    unsigned short* wf2T = (unsigned short*)alloc((size_t)FF_ * D_ * 2);
    float*          biasb= (float*)         alloc((size_t)B_ * KP_ * 4);
    float*          glog = (float*)         alloc(NE_ * 4);
    float*          wgt  = (float*)         alloc(NE_ * 4);
    float*          ck   = (float*)         alloc((size_t)P_ * D_ * 4);
    float*          cvt  = (float*)         alloc((size_t)P_ * D_ * 4);
    float*          cv   = (float*)         alloc((size_t)P_ * D_ * 4);
    float*          pooled=(float*)         alloc((size_t)B_ * D_ * 4);
    float*          moe  = (float*)         alloc(256);

    init_kernel<<<1, 32, 0, stream>>>(moe);

    // embeddings + LN
    {
        unsigned tot = (unsigned)(((size_t)M_ * D_ + 255) / 256);
        emb_kernel<<<tot, 256, 0, stream>>>(ids, wemb, pemb, temb, y);
        ln_row_kernel<<<M_, 256, 0, stream>>>(y, nullptr, nullptr, elng, elnb, x, xb);
    }
    bias_kernel<<<(B_ * KP_ + 255) / 256, 256, 0, stream>>>(amask, biasb);

    for (int l = 0; l < L_; ++l) {
        // bf16-transposed weights for this layer
        cvtT_kernel<<<(unsigned)(((size_t)D_ * QKV_ + 255) / 256), 256, 0, stream>>>(qkvw + (size_t)l * D_ * QKV_, wqT, D_, QKV_);
        cvtT_kernel<<<(unsigned)(((size_t)D_ * D_ + 255) / 256), 256, 0, stream>>>(aow + (size_t)l * D_ * D_, waoT, D_, D_);
        cvtT_kernel<<<(unsigned)(((size_t)D_ * FF_ + 255) / 256), 256, 0, stream>>>(f1w + (size_t)l * D_ * FF_, wf1T, D_, FF_);
        cvtT_kernel<<<(unsigned)(((size_t)FF_ * D_ + 255) / 256), 256, 0, stream>>>(f2w + (size_t)l * FF_ * D_, wf2T, FF_, D_);

        // MoE gate -> prefix K/V
        gate_kernel<<<NE_, 256, 0, stream>>>(EE, gw, gb, l, glog);
        gate_softmax_kernel<<<1, 32, 0, stream>>>(glog, wgt, moe);
        prefix_kernel<<<(P_ * D_ + 255) / 256, 256, 0, stream>>>(EE, wgt, l, ck, cvt);
        proj_kernel<<<(P_ * D_ + 255) / 256, 256, 0, stream>>>(cvt, pw, pbias, l, cv);
        prefix_write_kernel<<<(B_ * H_ * E_ * 32 + 255) / 256, 256, 0, stream>>>(ck, cv, Kf, Vt);

        // QKV projection (WMMA) + pack
        gemm_bf16_kernel<<<dim3(QKV_ / 64, M_ / 128), 256, 0, stream>>>(xb, wqT, y, QKV_, D_);
        pack_qkv_kernel<<<(unsigned)(((size_t)M_ * D_ + 255) / 256), 256, 0, stream>>>(y, qkvb, l, Qm, Kf, Vt);

        // attention (WMMA QK^T + softmax + WMMA PV)
        attn_kernel<<<dim3(S_ / 16, H_, B_), 32, 0, stream>>>(Qm, Kf, Vt, biasb, ctx);

        // attn output proj + residual LN1
        gemm_bf16_kernel<<<dim3(D_ / 64, M_ / 128), 256, 0, stream>>>(ctx, waoT, y, D_, D_);
        ln_row_kernel<<<M_, 256, 0, stream>>>(y, x, aob + (size_t)l * D_,
                                              ln1g + (size_t)l * D_, ln1b + (size_t)l * D_, x, xb);

        // FFN
        gemm_bf16_kernel<<<dim3(FF_ / 64, M_ / 128), 256, 0, stream>>>(xb, wf1T, y, FF_, D_);
        gelu_kernel<<<(unsigned)(((size_t)M_ * FF_ + 255) / 256), 256, 0, stream>>>(y, f1b, l, yb);
        gemm_bf16_kernel<<<dim3(D_ / 64, M_ / 128), 256, 0, stream>>>(yb, wf2T, y, D_, FF_);
        ln_row_kernel<<<M_, 256, 0, stream>>>(y, x, f2b + (size_t)l * D_,
                                              ln2g + (size_t)l * D_, ln2b + (size_t)l * D_, x, xb);
    }

    pool_kernel<<<(B_ * D_ + 255) / 256, 256, 0, stream>>>(x, poolw, poolb, pooled);
    logits_kernel<<<1, 64, 0, stream>>>(pooled, fcw, fcb, moe, out);
}
